// MultiHeadAttention_13391708029505
// MI455X (gfx1250) — matrix operand order repaired and checked by reading, not compiled, not run on hardware
//
#include <hip/hip_runtime.h>
#include <hip/hip_bf16.h>
#include <stdint.h>

// MI455X / gfx1250: wave32, WMMA 16x16x32 bf16 (fp32 accum), TDM K-tile staging.
// b=4, t=1024, d=512, h=8, kc=64, window=4.

typedef __attribute__((ext_vector_type(16))) __bf16 bf16x16;
typedef __attribute__((ext_vector_type(8)))  __bf16 bf16x8;
typedef __attribute__((ext_vector_type(8)))  float  f32x8;
typedef __attribute__((ext_vector_type(4)))  unsigned int u32x4;
typedef __attribute__((ext_vector_type(4)))  int    i32x4v;
typedef __attribute__((ext_vector_type(8)))  int    i32x8v;

#if defined(__has_builtin)
#if __has_builtin(__builtin_amdgcn_tensor_load_to_lds) && \
    __has_builtin(__builtin_amdgcn_s_wait_tensorcnt)
#define HAVE_TDM 1
#endif
#endif

#define DEV static __device__ __forceinline__

DEV uint16_t f2bf(float f) {
  union { float f; uint32_t u; } v; v.f = f;
  uint32_t u = v.u;
  u += 0x7FFFu + ((u >> 16) & 1u);      // round-to-nearest-even
  return (uint16_t)(u >> 16);
}
DEV float bf2f(uint16_t h) {
  union { uint32_t u; float f; } v; v.u = ((uint32_t)h) << 16;
  return v.f;
}
DEV bf16x16 cat8(bf16x8 a, bf16x8 b) {
  return __builtin_shufflevector(a, b, 0,1,2,3,4,5,6,7,8,9,10,11,12,13,14,15);
}
DEV f32x8 wmma_bf16(bf16x16 a, bf16x16 b, f32x8 c) {
  return __builtin_amdgcn_wmma_f32_16x16x32_bf16(false, a, false, b, (short)0, c,
                                                 false, false);
}

// ---------------------------------------------------------------------------
// Kernel 1: fused QKV projection. C[4096,512] = SRC[4096,512] @ W[512,512] + b
// grid (32, 8, 3): 128x64 tile per block, 8 waves, each wave 16x64 strip.
// Epilogue scatters through the faithful reshape [b,t,d]->[b,h,t,kc] (bf16).
// ---------------------------------------------------------------------------
__global__ __launch_bounds__(256)
void proj_qkv_kernel(const float* __restrict__ x, const float* __restrict__ c,
                     const float* __restrict__ wq, const float* __restrict__ bq,
                     const float* __restrict__ wk, const float* __restrict__ bk,
                     const float* __restrict__ wv, const float* __restrict__ bv,
                     uint16_t* __restrict__ qh, uint16_t* __restrict__ kh,
                     uint16_t* __restrict__ vh) {
  __shared__ __attribute__((aligned(16))) uint16_t atile[128][40]; // [m][k]
  __shared__ __attribute__((aligned(16))) uint16_t btile[64][40];  // [n][k] (transposed)

  const int tid  = threadIdx.x;
  const int lane = tid & 31;
  const int w    = tid >> 5;
  const int n16  = lane & 15;
  const int hi   = lane >> 4;
  const int bm = blockIdx.x, bn = blockIdx.y, z = blockIdx.z;

  const float* src  = (z == 0) ? x  : c;
  const float* wgt  = (z == 0) ? wq : (z == 1 ? wk : wv);
  const float* bias = (z == 0) ? bq : (z == 1 ? bk : bv);
  uint16_t*    dst  = (z == 0) ? qh : (z == 1 ? kh : vh);

  f32x8 acc[4];
#pragma unroll
  for (int nt = 0; nt < 4; ++nt) acc[nt] = (f32x8){0,0,0,0,0,0,0,0};

  for (int kk = 0; kk < 512; kk += 32) {
    // A tile: float4 global loads, packed 2x32b LDS stores
#pragma unroll
    for (int s = 0; s < 4; ++s) {
      int seg = tid + s * 256;                 // 1024 segments of 4 floats
      int ar = seg >> 3, ac4 = (seg & 7) * 4;
      float4 v = *(const float4*)(src + (size_t)(bm * 128 + ar) * 512 + kk + ac4);
      uint32_t lo = (uint32_t)f2bf(v.x) | ((uint32_t)f2bf(v.y) << 16);
      uint32_t hi2 = (uint32_t)f2bf(v.z) | ((uint32_t)f2bf(v.w) << 16);
      uint32_t* p = (uint32_t*)&atile[ar][ac4];
      p[0] = lo; p[1] = hi2;
    }
#pragma unroll
    for (int s = 0; s < 8; ++s) {
      int e = tid + s * 256;                   // 32x64 weights, transposed
      int kd = e >> 6, nc = e & 63;
      btile[nc][kd] = f2bf(wgt[(size_t)(kk + kd) * 512 + bn * 64 + nc]);
    }
    __syncthreads();

    // A frag: row = lane&15, K(e) = hi*16 + e
    const uint16_t* arow = &atile[w * 16 + n16][0];
    bf16x16 afrag = cat8(*(const bf16x8*)(arow + hi * 16),
                         *(const bf16x8*)(arow + hi * 16 + 8));
#pragma unroll
    for (int nt = 0; nt < 4; ++nt) {
      // B frag: col = lane&15, K(e) = hi*16 + e (contiguous in btile[n][.])
      const uint16_t* brow = &btile[nt * 16 + n16][0];
      bf16x16 bfrag = cat8(*(const bf16x8*)(brow + hi * 16),
                           *(const bf16x8*)(brow + hi * 16 + 8));
      acc[nt] = wmma_bf16(afrag, bfrag, acc[nt]);
    }
    __syncthreads();
  }

  // C layout: m = hi*8 + r, n = lane&15. Scatter with reshape permutation.
#pragma unroll
  for (int nt = 0; nt < 4; ++nt) {
    int ncol = bn * 64 + nt * 16 + n16;
    float bsv = bias[ncol];
#pragma unroll
    for (int r = 0; r < 8; ++r) {
      int m = hi * 8 + r;
      int g = bm * 128 + w * 16 + m;
      int b_ = g >> 10, time = g & 1023;
      float val = acc[nt][r] + bsv;
      uint32_t f = (uint32_t)time * 512u + (uint32_t)ncol;
      uint32_t hh = f >> 16, cidx = (f >> 10) & 63u, t2 = f & 1023u;
      dst[(((size_t)b_ * 8 + hh) * 1024 + t2) * 64 + cidx] = f2bf(val);
    }
  }
}

// ---------------------------------------------------------------------------
// Kernel 2: fused flash attention with windowed relative positions.
// grid (T/128=8, H=8, B=4); block 256 = 8 waves; wave owns 16 query rows.
// Online softmax over 16 key chunks of 64; T x T scores never hit memory.
// K chunk staged by the Tensor Data Mover (2-D tile, HW row padding to the
// LDS stride of 72 ushorts); V chunk transposed manually for the B layout.
// ---------------------------------------------------------------------------
__global__ __launch_bounds__(256)
void rel_attn_kernel(const uint16_t* __restrict__ qh, const uint16_t* __restrict__ kh,
                     const uint16_t* __restrict__ vh,
                     const float* __restrict__ emb_k, const float* __restrict__ emb_v,
                     uint16_t* __restrict__ oh) {
  __shared__ __attribute__((aligned(16))) uint16_t ktile[64][72]; // [s][kc]
  __shared__ __attribute__((aligned(16))) uint16_t vT[64][72];    // [kc][s]
  __shared__ __attribute__((aligned(16))) uint16_t pscr[8][16][72];
  __shared__ float rk[8][16][12];
  __shared__ float pband[8][16][12];
  __shared__ float ekl[9][64];
  __shared__ float evl[9][64];

  const int tid  = threadIdx.x;
  const int lane = tid & 31;
  const int w    = tid >> 5;
  const int n16  = lane & 15;
  const int hi   = lane >> 4;
  const int bh   = blockIdx.z * 8 + blockIdx.y;
  const size_t base = (size_t)bh << 16;            // * 1024 * 64
  const int tw = blockIdx.x * 128 + w * 16;        // wave's first query row

  for (int e = tid; e < 9 * 64; e += 256) {
    ekl[e >> 6][e & 63] = emb_k[e];
    evl[e >> 6][e & 63] = emb_v[e];
  }
  for (int e = tid; e < 8 * 16 * 12; e += 256) ((float*)pband)[e] = 0.f;
  __syncthreads();

  // rk[m][d] = q[tw+m] . emb_rel_k[d]
  for (int j = lane; j < 144; j += 32) {
    int m = j / 9, d = j % 9;
    const uint16_t* qrow = qh + base + (size_t)(tw + m) * 64;
    float s = 0.f;
    for (int kd = 0; kd < 64; ++kd) s += bf2f(qrow[kd]) * ekl[d][kd];
    rk[w][m][d] = s;
  }

  // Q A-fragments (kc 0..31 and 32..63), straight from global (L2-resident)
  const uint16_t* qr = qh + base + (size_t)(tw + n16) * 64;
  bf16x16 qa[2];
#pragma unroll
  for (int f = 0; f < 2; ++f)
    qa[f] = cat8(*(const bf16x8*)(qr + f * 32 + hi * 16),
                 *(const bf16x8*)(qr + f * 32 + hi * 16 + 8));

  float m_i[8], l_i[8];
  f32x8 acc[4];
#pragma unroll
  for (int r = 0; r < 8; ++r) { m_i[r] = -1e30f; l_i[r] = 0.f; }
#pragma unroll
  for (int nt = 0; nt < 4; ++nt) acc[nt] = (f32x8){0,0,0,0,0,0,0,0};
  __syncthreads();

  const float inv = 0.125f;                        // 1/sqrt(64)
  for (int kb = 0; kb < 16; ++kb) {
    const int kbase = kb * 64;
    if (kb + 1 < 16)
      __builtin_prefetch(vh + base + (size_t)(kbase + 64) * 64 + lane * 32, 0, 1);

#if HAVE_TDM
    // Tensor Data Mover: 2-D tile 64 rows x 128B, HW pad 4 DWORDs per 32-DWORD
    // row -> LDS stride 144B (= ktile[64][72]). Issued once per block (wave 0),
    // tracked with TENSORcnt; barrier below releases consumer waves.
    if (w == 0) {
      uint64_t ga = (uint64_t)(uintptr_t)(kh + base + (size_t)kbase * 64);
      u32x4 g0;
      g0.x = 1u;                                   // count=1, user descriptor
      g0.y = (uint32_t)(uintptr_t)&ktile[0][0];    // lds_addr = addr[31:0]
      g0.z = (uint32_t)ga;                         // global_addr[31:0]
      g0.w = (uint32_t)((ga >> 32) & 0x1FFFFFFu) | (2u << 30); // [56:32]|type=2
      i32x8v g1;
      g1[0] = (int)((1u << 16)       // data_size = 2 bytes
                  | (1u << 20)       // pad_enable
                  | (4u << 22)       // pad_interval: 32 DWORDs (one 128B row)
                  | (3u << 25));     // pad_amount: 4 DWORDs (16B)
      g1[1] = (int)(64u << 16);      // tensor_dim0 = 64 elems (low 16)
      g1[2] = (int)(1024u << 16);    // tensor_dim1 = 1024 rows (low 16)
      g1[3] = (int)(64u << 16);      // tile_dim0 = 64
      g1[4] = 64;                    // tile_dim1 = 64, tile_dim2 = 0
      g1[5] = 64;                    // tensor_dim0_stride = 64 elems
      g1[6] = 0;
      g1[7] = 0;
      i32x4v gz = {0, 0, 0, 0};
#if __clang_major__ >= 23
      i32x8v gz8 = {0, 0, 0, 0, 0, 0, 0, 0};
      __builtin_amdgcn_tensor_load_to_lds(g0, g1, gz, gz, gz8, 0);
#else
      __builtin_amdgcn_tensor_load_to_lds(g0, g1, gz, gz, 0);
#endif
      __builtin_amdgcn_s_wait_tensorcnt(0);
    }
#else
#pragma unroll
    for (int s = 0; s < 2; ++s) {                  // fallback: vector copies
      int e = tid + s * 256;
      int srel = e >> 3, k8 = (e & 7) * 8;
      *(bf16x8*)&ktile[srel][k8] =
          *(const bf16x8*)(kh + base + (size_t)(kbase + srel) * 64 + k8);
    }
#endif

    // V chunk: vector global loads, scalar transposed LDS stores
#pragma unroll
    for (int s = 0; s < 2; ++s) {
      int e = tid + s * 256;                       // 512 segments of 8
      int key = e >> 3, c8 = (e & 7) * 8;
      bf16x8 v = *(const bf16x8*)(vh + base + (size_t)(kbase + key) * 64 + c8);
#pragma unroll
      for (int j = 0; j < 8; ++j)
        ((__bf16*)&vT[c8 + j][0])[key] = v[j];
    }
    __syncthreads();

    // S = Q @ K^T  (16 x 64 per wave)
    f32x8 s[4];
#pragma unroll
    for (int nt = 0; nt < 4; ++nt) {
      s[nt] = (f32x8){0,0,0,0,0,0,0,0};
      const uint16_t* krow = &ktile[nt * 16 + n16][0];
#pragma unroll
      for (int ki = 0; ki < 2; ++ki) {
        bf16x16 bfrag = cat8(*(const bf16x8*)(krow + ki * 32 + hi * 16),
                             *(const bf16x8*)(krow + ki * 32 + hi * 16 + 8));
        s[nt] = wmma_bf16(qa[ki], bfrag, s[nt]);
      }
    }

    // banded relative-key term + scale
    const bool band = (kbase <= tw + 19) && (kbase + 63 >= tw - 4);
#pragma unroll
    for (int nt = 0; nt < 4; ++nt)
#pragma unroll
      for (int r = 0; r < 8; ++r) {
        float v = s[nt][r];
        if (band) {
          int m = hi * 8 + r;
          int d = (kbase + nt * 16 + n16) - (tw + m) + 4;
          if ((unsigned)d <= 8u) v += rk[w][m][d];
        }
        s[nt][r] = v * inv;
      }

    // online softmax: row max over the 16 lanes of each half
    float al[8];
#pragma unroll
    for (int r = 0; r < 8; ++r) {
      float v = fmaxf(fmaxf(s[0][r], s[1][r]), fmaxf(s[2][r], s[3][r]));
#pragma unroll
      for (int msk = 1; msk < 16; msk <<= 1) v = fmaxf(v, __shfl_xor(v, msk, 32));
      float mn = fmaxf(m_i[r], v);
      al[r] = __expf(m_i[r] - mn);
      m_i[r] = mn;
    }
    if (n16 < 9)
#pragma unroll
      for (int r = 0; r < 8; ++r) pband[w][hi * 8 + r][n16] *= al[r];
#pragma unroll
    for (int r = 0; r < 8; ++r) l_i[r] *= al[r];
#pragma unroll
    for (int nt = 0; nt < 4; ++nt)
#pragma unroll
      for (int r = 0; r < 8; ++r) acc[nt][r] *= al[r];

    float rs[8] = {0, 0, 0, 0, 0, 0, 0, 0};
#pragma unroll
    for (int nt = 0; nt < 4; ++nt)
#pragma unroll
      for (int r = 0; r < 8; ++r) {
        float p = __expf(s[nt][r] - m_i[r]);
        s[nt][r] = p;
        rs[r] += p;
      }
#pragma unroll
    for (int r = 0; r < 8; ++r) {
      float v = rs[r];
#pragma unroll
      for (int msk = 1; msk < 16; msk <<= 1) v += __shfl_xor(v, msk, 32);
      l_i[r] += v;
    }

    // P to LDS scratch (C-layout -> A-layout crossing) + value-band weights
#pragma unroll
    for (int nt = 0; nt < 4; ++nt)
#pragma unroll
      for (int r = 0; r < 8; ++r) {
        int m = hi * 8 + r;
        pscr[w][m][nt * 16 + n16] = f2bf(s[nt][r]);
        if (band) {
          int d = (kbase + nt * 16 + n16) - (tw + m) + 4;
          if ((unsigned)d <= 8u) pband[w][m][d] += s[nt][r];
        }
      }

    // O += P @ V
    const uint16_t* prow = &pscr[w][n16][0];
#pragma unroll
    for (int ki = 0; ki < 2; ++ki) {
      bf16x16 pf = cat8(*(const bf16x8*)(prow + ki * 32 + hi * 16),
                        *(const bf16x8*)(prow + ki * 32 + hi * 16 + 8));
#pragma unroll
      for (int nt = 0; nt < 4; ++nt) {
        const uint16_t* vrow = &vT[nt * 16 + n16][0];
        bf16x16 bfrag = cat8(*(const bf16x8*)(vrow + ki * 32 + hi * 16),
                             *(const bf16x8*)(vrow + ki * 32 + hi * 16 + 8));
        acc[nt] = wmma_bf16(pf, bfrag, acc[nt]);
      }
    }
    __syncthreads();
  }

  // epilogue: add relative-value band, normalize, store bf16 head output
#pragma unroll
  for (int nt = 0; nt < 4; ++nt)
#pragma unroll
    for (int r = 0; r < 8; ++r) {
      int m = hi * 8 + r, cc = nt * 16 + n16;
      float v = acc[nt][r];
      float badd = 0.f;
#pragma unroll
      for (int d = 0; d < 9; ++d) badd += pband[w][m][d] * evl[d][cc];
      v = (v + badd) / l_i[r];
      oh[base + (size_t)(tw + m) * 64 + cc] = f2bf(v);
    }
}

// ---------------------------------------------------------------------------
// Kernel 3: output projection with inverse reshape gather. fp32 out + bias.
// ---------------------------------------------------------------------------
__global__ __launch_bounds__(256)
void out_proj_kernel(const uint16_t* __restrict__ oh, const float* __restrict__ wo,
                     const float* __restrict__ bo, float* __restrict__ out) {
  __shared__ __attribute__((aligned(16))) uint16_t atile[128][40];
  __shared__ __attribute__((aligned(16))) uint16_t btile[64][40];

  const int tid  = threadIdx.x;
  const int lane = tid & 31;
  const int w    = tid >> 5;
  const int n16  = lane & 15;
  const int hi   = lane >> 4;
  const int bm = blockIdx.x, bn = blockIdx.y;

  f32x8 acc[4];
#pragma unroll
  for (int nt = 0; nt < 4; ++nt) acc[nt] = (f32x8){0,0,0,0,0,0,0,0};

  for (int kk = 0; kk < 512; kk += 32) {
#pragma unroll
    for (int s = 0; s < 16; ++s) {
      int e = tid + s * 256;
      int ar = e >> 5, ac = e & 31;
      int g = bm * 128 + ar;
      int b_ = g >> 10, time = g & 1023;
      uint32_t f = (uint32_t)time * 512u + (uint32_t)(kk + ac);
      uint32_t hh = f >> 16, cidx = (f >> 10) & 63u, t2 = f & 1023u;
      atile[ar][ac] = oh[(((size_t)b_ * 8 + hh) * 1024 + t2) * 64 + cidx];
    }
#pragma unroll
    for (int s = 0; s < 8; ++s) {
      int e = tid + s * 256;
      int kd = e >> 6, nc = e & 63;
      btile[nc][kd] = f2bf(wo[(size_t)(kk + kd) * 512 + bn * 64 + nc]);
    }
    __syncthreads();

    const uint16_t* arow = &atile[w * 16 + n16][0];
    bf16x16 afrag = cat8(*(const bf16x8*)(arow + hi * 16),
                         *(const bf16x8*)(arow + hi * 16 + 8));
#pragma unroll
    for (int nt = 0; nt < 4; ++nt) {
      const uint16_t* brow = &btile[nt * 16 + n16][0];
      bf16x16 bfrag = cat8(*(const bf16x8*)(brow + hi * 16),
                           *(const bf16x8*)(brow + hi * 16 + 8));
      acc[nt] = wmma_bf16(afrag, bfrag, acc[nt]);
    }
    __syncthreads();
  }

#pragma unroll
  for (int nt = 0; nt < 4; ++nt) {
    int ncol = bn * 64 + nt * 16 + n16;
    float bsv = bo[ncol];
#pragma unroll
    for (int r = 0; r < 8; ++r) {
      int g = bm * 128 + w * 16 + hi * 8 + r;
      out[(size_t)g * 512 + ncol] = acc[nt][r] + bsv;
    }
  }
}

extern "C" void kernel_launch(void* const* d_in, const int* in_sizes, int n_in,
                              void* d_out, int out_size, void* d_ws, size_t ws_size,
                              hipStream_t stream) {
  const float* x  = (const float*)d_in[0];
  const float* c  = (const float*)d_in[1];
  const float* wq = (const float*)d_in[2];
  const float* bq = (const float*)d_in[3];
  const float* wk = (const float*)d_in[4];
  const float* bk = (const float*)d_in[5];
  const float* wv = (const float*)d_in[6];
  const float* bv = (const float*)d_in[7];
  const float* wo = (const float*)d_in[8];
  const float* bo = (const float*)d_in[9];
  const float* ek = (const float*)d_in[10];
  const float* ev = (const float*)d_in[11];
  float* out = (float*)d_out;

  const size_t per = (size_t)4 * 8 * 1024 * 64;   // 2M elems per head tensor
  uint16_t* qh = (uint16_t*)d_ws;                  // bf16 [b,h,t,kc]
  uint16_t* kh = qh + per;
  uint16_t* vh = kh + per;
  uint16_t* oh = vh + per;                         // total 16 MB of d_ws

  dim3 blk(256);
  proj_qkv_kernel<<<dim3(32, 8, 3), blk, 0, stream>>>(x, c, wq, bq, wk, bk,
                                                      wv, bv, qh, kh, vh);
  rel_attn_kernel<<<dim3(8, 8, 4), blk, 0, stream>>>(qh, kh, vh, ek, ev, oh);
  out_proj_kernel<<<dim3(32, 8), blk, 0, stream>>>(oh, wo, bo, out);
}
